// Model_73950746902846
// MI455X (gfx1250) — matrix operand order repaired and checked by reading, not compiled, not run on hardware
//
#include <hip/hip_runtime.h>
#include <hip/hip_bf16.h>

// ---------------- problem constants ----------------
#define T_STEPS 2144          // SEQ_LEN + PRED_LEN
#define SEQ_LEN 2048
#define PRED_LEN 96
#define BATCH   512
#define HID     40
#define G4      160           // 4*HID
#define BT      16            // batch tile per workgroup
#define NWAVE   10            // 10 waves -> 160 gate columns
#define NTHR    (NWAVE * 32)  // 320 threads
#define TB      ((size_t)T_STEPS * BATCH)

// LDS layout (bytes)
#define OFF_WT0 0u            // f16 [160][64]  transposed W for layer0 (x|h1 -> gates)
#define OFF_WT1 20480u        // f16 [160][96]  transposed W for layer1 (h1|h2 -> gates)
#define OFF_WTH 51200u        // f16 [32][96]   transposed head W (hp -> beta|gamma)
#define OFF_A0  57344u        // f16 [16][64]   layer0 input rows [x_t(9) h1(40) pad]
#define OFF_A1  59392u        // f16 [16][96]   layer1 input rows [h1(40) h2(40) pad]
#define OFF_AHP 62464u        // f16 [16][96]   interleaved hp rows (pad to 96)
#define OFF_G   65536u        // f32 [16][160]  gate pre-activations
#define OFF_C1  75776u        // f32 [16][40]
#define OFF_C2  78336u        // f32 [16][40]
#define OFF_B0  80896u        // f32 [160] bias layer0 (b_ih+b_hh)
#define OFF_B1  81536u        // f32 [160] bias layer1
#define OFF_BH  82176u        // f32 [32]  head bias
#define LDS_BYTES 82304u

typedef __attribute__((ext_vector_type(16))) _Float16 v16h;
typedef __attribute__((ext_vector_type(8)))  float    v8f;

struct alignas(16) H8 { _Float16 h[8]; };
union AF { v16h v; H8 q[2]; };

// fast activations: single v_rcp_f32 / v_exp_f32 / v_log_f32, no IEEE-div expansion
__device__ __forceinline__ float frcp(float x)  { return __builtin_amdgcn_rcpf(x); }
__device__ __forceinline__ float sigf(float x)  { return frcp(1.0f + __expf(-x)); }
__device__ __forceinline__ float ftanh(float x) { return 1.0f - 2.0f * frcp(__expf(2.0f * x) + 1.0f); }
__device__ __forceinline__ float softplusf(float x) {
    return (x > 15.0f) ? x : __logf(1.0f + __expf(x));
}

// One 16x16 output tile: A[16 x 32*KT] (row pitch apitch halves) times
// Wt (stored N-major: row n holds K contiguous, pitch wpitch halves).
// Fragment packing follows CDNA5 ISA 7.12.2 16-bit A (16x32) and B (32x16) layouts.
template<int KT>
__device__ __forceinline__ v8f wmma_block(const _Float16* A, int apitch,
                                          const _Float16* Wt, int wpitch,
                                          int n0, int lane) {
    const int m  = lane & 15;   // A row (M) and B column (N) sub-index
    const int hi = lane >> 4;   // half-wave select
    v8f acc = {};
#pragma unroll
    for (int kt = 0; kt < KT; ++kt) {
        const int k0 = kt * 32;
        AF a, b;
        // A lanes 0-15: e0..7 -> K 0..7,  e8..15 -> K 16..23
        // A lanes16-31: e0..7 -> K 8..15, e8..15 -> K 24..31
        a.q[0] = *(const H8*)(A + m * apitch + k0 + hi * 8);
        a.q[1] = *(const H8*)(A + m * apitch + k0 + 16 + hi * 8);
        // B lanes 0-15: e0..7 -> K 0..7, e8..15 -> K 16..23; lanes 16-31: e0..7 -> K 8..15, e8..15 -> K 24..31 (column n0+m)
        b.q[0] = *(const H8*)(Wt + (n0 + m) * wpitch + k0 + hi * 8);
        b.q[1] = *(const H8*)(Wt + (n0 + m) * wpitch + k0 + 16 + hi * 8);
        acc = __builtin_amdgcn_wmma_f32_16x16x32_f16(
            /*neg_a=*/false, a.v, /*neg_b=*/false, b.v,
            /*c_mod=*/(short)0, acc, /*reuse_a=*/false, /*reuse_b=*/false);
    }
    return acc;
}

__global__ __launch_bounds__(NTHR)
void lstm_fused_kernel(const float* __restrict__ x_enc,   // [512,2048,10]
                       const float* __restrict__ y_enc,   // [512,96,10]
                       const float* __restrict__ Wih0, const float* __restrict__ Whh0,
                       const float* __restrict__ bih0, const float* __restrict__ bhh0,
                       const float* __restrict__ Wih1, const float* __restrict__ Whh1,
                       const float* __restrict__ bih1, const float* __restrict__ bhh1,
                       const float* __restrict__ Wb,   const float* __restrict__ bb,
                       const float* __restrict__ Wg,   const float* __restrict__ bgm,
                       float* __restrict__ out) {
    extern __shared__ char smem[];
    _Float16* Wt0 = (_Float16*)(smem + OFF_WT0);
    _Float16* Wt1 = (_Float16*)(smem + OFF_WT1);
    _Float16* Wth = (_Float16*)(smem + OFF_WTH);
    _Float16* A0  = (_Float16*)(smem + OFF_A0);
    _Float16* A1  = (_Float16*)(smem + OFF_A1);
    _Float16* Ahp = (_Float16*)(smem + OFF_AHP);
    float* Gm  = (float*)(smem + OFF_G);
    float* c1s = (float*)(smem + OFF_C1);
    float* c2s = (float*)(smem + OFF_C2);
    float* b0  = (float*)(smem + OFF_B0);
    float* b1  = (float*)(smem + OFF_B1);
    float* bh  = (float*)(smem + OFF_BH);

    const int tid  = threadIdx.x;
    const int w    = tid >> 5;          // wave id 0..9
    const int lane = tid & 31;
    const int bt   = blockIdx.x * BT;   // global batch tile base

    // ---- stage transposed f16 weights + fused biases in LDS ----
    for (int i = tid; i < G4 * 64; i += NTHR) {          // layer0: K = [x(9) h1(40)] pad 64
        int n = i >> 6, k = i & 63;
        float v = 0.0f;
        if (k < 9)       v = Wih0[n * 9 + k];
        else if (k < 49) v = Whh0[n * HID + (k - 9)];
        Wt0[n * 64 + k] = (_Float16)v;
    }
    for (int i = tid; i < G4 * 96; i += NTHR) {          // layer1: K = [h1(40) h2(40)] pad 96
        int n = i / 96, k = i % 96;
        float v = 0.0f;
        if (k < 40)      v = Wih1[n * HID + k];
        else if (k < 80) v = Whh1[n * HID + (k - 40)];
        Wt1[n * 96 + k] = (_Float16)v;
    }
    for (int i = tid; i < 32 * 96; i += NTHR) {          // head: n0=beta, n1..20=gamma
        int n = i / 96, k = i % 96;
        float v = 0.0f;
        if (k < 80) {
            if (n == 0)       v = Wb[k];
            else if (n <= 20) v = Wg[(n - 1) * 80 + k];
        }
        Wth[n * 96 + k] = (_Float16)v;
    }
    for (int j = tid; j < G4; j += NTHR) { b0[j] = bih0[j] + bhh0[j]; b1[j] = bih1[j] + bhh1[j]; }
    for (int j = tid; j < 32; j += NTHR) bh[j] = (j == 0) ? bb[0] : ((j <= 20) ? bgm[j - 1] : 0.0f);
    // zero recurrent state + activation pads
    for (int i = tid; i < BT * 64; i += NTHR) A0[i] = (_Float16)0.0f;
    for (int i = tid; i < BT * 96; i += NTHR) { A1[i] = (_Float16)0.0f; Ahp[i] = (_Float16)0.0f; }
    for (int i = tid; i < BT * HID; i += NTHR) { c1s[i] = 0.0f; c2s[i] = 0.0f; }
    __syncthreads();

    // ---------------- hoisted per-thread invariants ----------------
    // phase A: one (batch,row-feature) slot per thread, running src pointer,
    // x value for the *current* step pre-loaded (software pipeline).
    const bool ld_act = (tid < BT * 10);
    const int  ld_b   = tid / 10, ld_k = tid % 10;
    const int  ld_bg  = bt + ld_b;
    const float* ld_src = x_enc + ((size_t)ld_bg * SEQ_LEN) * 10 + ld_k;
    float  xv = 0.0f;
    float* lbl_ptr = out + 21 * TB + ld_bg;   // advance by BATCH per step
    if (ld_act) xv = *ld_src;                 // t = 0 value
    // after use at step t we load for step t+1; pointer below always targets t+1
    const float* ld_nxt = ld_src + 10;

    // phases C/E: two fixed (b,j) items per thread; hoist bias values
    int   ew_goff[2], ew_coff[2], ew_a0off[2], ew_a1off[2], ew_hpoff[2];
    float bw0[2][4], bw1[2][4];
#pragma unroll
    for (int it = 0; it < 2; ++it) {
        int item = tid + it * NTHR;
        int b = item / HID, j = item % HID;
        ew_goff[it]  = b * G4 + j;
        ew_coff[it]  = b * HID + j;
        ew_a0off[it] = b * 64 + 9 + j;
        ew_a1off[it] = b * 96 + j;
        ew_hpoff[it] = b * 96 + 2 * j;
        bw0[it][0] = b0[j];  bw0[it][1] = b0[HID + j];
        bw0[it][2] = b0[2 * HID + j]; bw0[it][3] = b0[3 * HID + j];
        bw1[it][0] = b1[j];  bw1[it][1] = b1[HID + j];
        bw1[it][2] = b1[2 * HID + j]; bw1[it][3] = b1[3 * HID + j];
    }

    // phase F: head column / row-base / bias, running offsets
    const int  hd_n   = w * 16 + (lane & 15);
    const int  hd_mb  = (lane >> 4) * 8;
    const bool hd_act   = (w < 2);
    const bool hd_beta  = hd_act && (hd_n == 0);
    const bool hd_gamma = hd_act && (hd_n >= 1) && (hd_n <= 20);
    const float hd_bias = hd_act ? bh[hd_n] : 0.0f;
    size_t hd_row  = (size_t)(bt + hd_mb);                       // row = t*BATCH + bg
    size_t hd_grow = TB + (size_t)(bt + hd_mb) * 20 + (hd_n - 1); // gamma base

    const int n0 = w * 16;
    const int mb = hd_mb;

    for (int t = 0; t < T_STEPS; ++t) {
        // ---- phase A: commit pre-loaded x_t, then issue load for t+1 ----
        if (ld_act) {
            if (ld_k < 9) A0[ld_b * 64 + ld_k] = (_Float16)xv;
            else          *lbl_ptr = xv;                 // labels [T,B]
            lbl_ptr += BATCH;
            if (t + 1 < T_STEPS) {
                const float* p = (t + 1 == SEQ_LEN)
                    ? (y_enc + ((size_t)ld_bg * PRED_LEN) * 10 + ld_k)
                    : ld_nxt;
                xv = *p;                                 // latency hidden behind GEMMs
                ld_nxt = p + 10;
            }
        }
        __syncthreads();

        // ---- phase B: layer0 GEMM (each wave: one 16-col gate tile) ----
        {
            v8f acc = wmma_block<2>(A0, 64, Wt0, 64, n0, lane);
            const int n = n0 + (lane & 15);
#pragma unroll
            for (int r = 0; r < 8; ++r) Gm[(mb + r) * G4 + n] = acc[r];   // C/D: M=8*hi+r
        }
        __syncthreads();

        // ---- phase C: layer0 gates -> h1, c1 ----
#pragma unroll
        for (int it = 0; it < 2; ++it) {
            const float* g = Gm + ew_goff[it];
            float gi = sigf (g[0]       + bw0[it][0]);
            float gf = sigf (g[HID]     + bw0[it][1]);
            float gg = ftanh(g[2 * HID] + bw0[it][2]);
            float go = sigf (g[3 * HID] + bw0[it][3]);
            float c = gf * c1s[ew_coff[it]] + gi * gg;
            c1s[ew_coff[it]] = c;
            _Float16 hh = (_Float16)(go * ftanh(c));
            A0[ew_a0off[it]]  = hh;    // recurrent input for layer0 (next step)
            A1[ew_a1off[it]]  = hh;    // feed-forward input for layer1 (this step)
            Ahp[ew_hpoff[it]] = hh;    // interleaved hp even slots
        }
        __syncthreads();

        // ---- phase D: layer1 GEMM ----
        {
            v8f acc = wmma_block<3>(A1, 96, Wt1, 96, n0, lane);
            const int n = n0 + (lane & 15);
#pragma unroll
            for (int r = 0; r < 8; ++r) Gm[(mb + r) * G4 + n] = acc[r];
        }
        __syncthreads();

        // ---- phase E: layer1 gates -> h2, c2 ----
#pragma unroll
        for (int it = 0; it < 2; ++it) {
            const float* g = Gm + ew_goff[it];
            float gi = sigf (g[0]       + bw1[it][0]);
            float gf = sigf (g[HID]     + bw1[it][1]);
            float gg = ftanh(g[2 * HID] + bw1[it][2]);
            float go = sigf (g[3 * HID] + bw1[it][3]);
            float c = gf * c2s[ew_coff[it]] + gi * gg;
            c2s[ew_coff[it]] = c;
            _Float16 hh = (_Float16)(go * ftanh(c));
            A1[ew_a1off[it] + HID] = hh;   // recurrent input for layer1 (next step)
            Ahp[ew_hpoff[it] + 1]  = hh;   // interleaved hp odd slots
        }
        __syncthreads();

        // ---- phase F: heads (waves 0,1) -> softplus -> global ----
        if (hd_act) {
            v8f acc = wmma_block<3>(Ahp, 96, Wth, 96, n0, lane);
#pragma unroll
            for (int r = 0; r < 8; ++r) {
                float sp = softplusf(acc[r] + hd_bias);
                if (hd_beta)       out[hd_row + r] = sp;            // beta_0 [T,B,1]
                else if (hd_gamma) out[hd_grow + (size_t)r * 20] = sp; // gamma [T,B,20]
            }
        }
        hd_row  += BATCH;
        hd_grow += (size_t)BATCH * 20;
        // no extra barrier: phase F touches Ahp/Wth/global only; next phase A
        // touches A0[:, :9]/labels only, and the end-of-A barrier orders both
        // before the next GEMM.
    }
}

extern "C" void kernel_launch(void* const* d_in, const int* in_sizes, int n_in,
                              void* d_out, int out_size, void* d_ws, size_t ws_size,
                              hipStream_t stream) {
    (void)in_sizes; (void)n_in; (void)out_size; (void)d_ws; (void)ws_size;
    const float* x_enc = (const float*)d_in[0];
    const float* y_enc = (const float*)d_in[3];
    const float* Wih0  = (const float*)d_in[5];
    const float* Whh0  = (const float*)d_in[6];
    const float* bih0  = (const float*)d_in[7];
    const float* bhh0  = (const float*)d_in[8];
    const float* Wih1  = (const float*)d_in[9];
    const float* Whh1  = (const float*)d_in[10];
    const float* bih1  = (const float*)d_in[11];
    const float* bhh1  = (const float*)d_in[12];
    const float* Wb    = (const float*)d_in[13];
    const float* bb    = (const float*)d_in[14];
    const float* Wg    = (const float*)d_in[15];
    const float* bgm   = (const float*)d_in[16];
    float* out = (float*)d_out;

    dim3 grid(BATCH / BT);   // 32 workgroups, one per 16-row batch tile
    dim3 block(NTHR);        // 10 wave32 waves
    lstm_fused_kernel<<<grid, block, LDS_BYTES, stream>>>(
        x_enc, y_enc, Wih0, Whh0, bih0, bhh0, Wih1, Whh1, bih1, bhh1,
        Wb, bb, Wg, bgm, out);
}
